// CausalSelfAttention_25761213841949
// MI455X (gfx1250) — matrix operand order repaired and checked by reading, not compiled, not run on hardware
//
#include <hip/hip_runtime.h>
#include <cstdint>
#include <cstddef>

// ---------------------------------------------------------------------------
// Problem constants (from reference): B=2, T=2048, C=512, H=8, D=64,
// block-causal mask with frame size 4 (NUM_ANIMALS).
// ---------------------------------------------------------------------------
#define BB   2
#define TT   2048
#define CC   512
#define HH   8
#define DD   64
#define MM   (BB * TT)          // 4096 rows in the token dimension

typedef unsigned short u16;
typedef __attribute__((ext_vector_type(16))) __bf16 v16bf;
typedef __attribute__((ext_vector_type(8)))  float  v8f;
typedef __attribute__((ext_vector_type(4)))  unsigned int v4u;

union Frag { v16bf v; v4u u[2]; };

// float -> bf16 (round-to-nearest-even)
__device__ __forceinline__ u16 f2bf(float f) {
    unsigned u = __float_as_uint(f);
    return (u16)((u + 0x7FFFu + ((u >> 16) & 1u)) >> 16);
}

__device__ __forceinline__ v8f zero8() {
    v8f z;
#pragma unroll
    for (int i = 0; i < 8; ++i) z[i] = 0.0f;
    return z;
}

__device__ __forceinline__ v8f wmma_bf16(v16bf a, v16bf b, v8f c) {
    // D = A(16x32 bf16) * B(32x16 bf16) + C(16x16 f32)
    return __builtin_amdgcn_wmma_f32_16x16x32_bf16(
        /*neg_a=*/false, a, /*neg_b=*/false, b,
        /*c_mod=*/(short)0, c, /*reuse_a=*/false, /*reuse_b=*/false);
}

// A fragment (16x32, 16-bit): lane holds row M=lane&15.
// elements e=0..7  -> K = kbase + half*8 + e        (contiguous 8 bf16)
// elements e=8..15 -> K = kbase + 16 + half*8 + e-8 (contiguous 8 bf16)
__device__ __forceinline__ v16bf load_afrag(const u16* p) {
    Frag f;
    f.u[0] = *reinterpret_cast<const v4u*>(p);
    f.u[1] = *reinterpret_cast<const v4u*>(p + 16);
    return f.v;
}

// B fragment (32x16, 16-bit): lane holds column N=lane&15.
// elements e=0..7  -> K = kbase + half*8 + e        (contiguous 8 bf16)
// elements e=8..15 -> K = kbase + 16 + half*8 + e-8 (contiguous 8 bf16)
__device__ __forceinline__ v16bf load_bfrag(const u16* p) {
    Frag f;
    f.u[0] = *reinterpret_cast<const v4u*>(p);
    f.u[1] = *reinterpret_cast<const v4u*>(p + 16);
    return f.v;
}

// async global -> LDS copy of 16 bytes per lane (CDNA5, tracked by ASYNCcnt)
__device__ __forceinline__ void async_ld_b128(unsigned lds_off, const u16* gptr) {
    asm volatile("global_load_async_to_lds_b128 %0, %1, off"
                 :: "v"(lds_off), "v"((unsigned long long)(uintptr_t)gptr)
                 : "memory");
}

// reductions within each 16-lane half of the wave32
__device__ __forceinline__ float redmax16(float v) {
#pragma unroll
    for (int off = 1; off < 16; off <<= 1) v = fmaxf(v, __shfl_xor(v, off, 32));
    return v;
}
__device__ __forceinline__ float redsum16(float v) {
#pragma unroll
    for (int off = 1; off < 16; off <<= 1) v += __shfl_xor(v, off, 32);
    return v;
}

// ---------------------------------------------------------------------------
// Kernel 1: f32 -> bf16 elementwise
// ---------------------------------------------------------------------------
__global__ __launch_bounds__(256) void csa_cvt_bf16(const float* __restrict__ in,
                                                    u16* __restrict__ out, int n) {
    int i = blockIdx.x * blockDim.x + threadIdx.x;
    if (i < n) out[i] = f2bf(in[i]);
}

// ---------------------------------------------------------------------------
// Kernel 2: W[CxC] f32 -> W^T[CxC] bf16  (out[n*C + k] = W[k*C + n])
// ---------------------------------------------------------------------------
__global__ __launch_bounds__(256) void csa_wT_bf16(const float* __restrict__ W,
                                                   u16* __restrict__ WT) {
    int idx = blockIdx.x * blockDim.x + threadIdx.x;   // 0 .. C*C-1
    int n = idx >> 9;      // /512
    int k = idx & (CC - 1);
    WT[idx] = f2bf(W[(size_t)k * CC + n]);
}

// ---------------------------------------------------------------------------
// Kernel 3: GEMM  out = A[M x C] (bf16) * W + bias, W given pre-transposed.
//   One wave per 64(M) x 64(N) tile; K swept in steps of 32.
//   Weight tile (64 cols x 32 k = 4KB) is staged global->LDS with
//   global_load_async_to_lds_b128, double buffered (ASYNCcnt pipelining);
//   A fragments load straight from global. 16 WMMAs per K-step.
//   mode 0: store bf16 as [B,H,T,D]   (Q, K)
//   mode 1: store bf16 as [B,H,D,T]   (V transposed for PV WMMA)
//   mode 2: store f32  as [M, C]      (final output projection)
// ---------------------------------------------------------------------------
__global__ __launch_bounds__(32) void csa_gemm64(const u16* __restrict__ A,
                                                 const u16* __restrict__ WT,
                                                 const float* __restrict__ bias,
                                                 void* __restrict__ outp,
                                                 int mode) {
    // two 4KB buffers: [buf][col(64)][k(32)] of bf16, col-major rows, k contiguous
    __shared__ alignas(16) u16 ldsB[2][64 * 32];

    const int lane = threadIdx.x;
    const int hf   = lane >> 4;
    const int l15  = lane & 15;
    const int i0   = blockIdx.x * 64;    // row block (64 rows)
    const int nb   = blockIdx.y * 64;    // column block (64 cols)

    const unsigned ldsBase = (unsigned)(uintptr_t)(&ldsB[0][0]);

    v8f acc[4][4];
#pragma unroll
    for (int g = 0; g < 4; ++g)
#pragma unroll
        for (int jt = 0; jt < 4; ++jt) acc[g][jt] = zero8();

    const u16* arow[4];
#pragma unroll
    for (int g = 0; g < 4; ++g)
        arow[g] = A + (size_t)(i0 + g * 16 + l15) * CC;

    // ---- stage one 64x32 weight tile into LDS buffer `buf` (8 x b128 / lane)
    auto stage = [&](int buf, int kk) {
#pragma unroll
        for (int ii = 0; ii < 8; ++ii) {
            const int ch  = lane + ii * 32;    // 256 chunks of 16B
            const int row = ch >> 2;           // 0..63 (output col)
            const int q   = ch & 3;            // 16B chunk within the 64B row
            async_ld_b128(ldsBase + (unsigned)(buf * 4096 + row * 64 + q * 16),
                          WT + (size_t)(nb + row) * CC + kk + q * 8);
        }
    };

    stage(0, 0);                 // prologue: 8 async ops in flight

    for (int it = 0; it < CC / 32; ++it) {
        const int kk  = it * 32;
        const int cur = it & 1;

        if (it + 1 < CC / 32) {
            // retire this wave's DS reads of the buffer we are about to refill
            asm volatile("s_wait_dscnt 0" ::: "memory");
            stage(cur ^ 1, kk + 32);                    // now 16 in flight
            asm volatile("s_wait_asynccnt 8" ::: "memory");  // first 8 done
        } else {
            asm volatile("s_wait_asynccnt 0" ::: "memory");
        }

        v16bf af[4];
#pragma unroll
        for (int g = 0; g < 4; ++g)
            af[g] = load_afrag(arow[g] + kk + hf * 8);

#pragma unroll
        for (int jt = 0; jt < 4; ++jt) {
            const u16* bp = &ldsB[cur][0] + (jt * 16 + l15) * 32 + hf * 8;
            v16bf bf = load_bfrag(bp);
#pragma unroll
            for (int g = 0; g < 4; ++g)
                acc[g][jt] = wmma_bf16(af[g], bf, acc[g][jt]);
        }
    }

    // C/D layout: lane holds column n = l15; vgpr r holds row m = hf*8 + r.
#pragma unroll
    for (int jt = 0; jt < 4; ++jt) {
        const int n = nb + jt * 16 + l15;
        const float bv = bias[n];
#pragma unroll
        for (int g = 0; g < 4; ++g) {
#pragma unroll
            for (int r = 0; r < 8; ++r) {
                const int i = i0 + g * 16 + hf * 8 + r;  // global row (b*T + t)
                const float val = acc[g][jt][r] + bv;
                if (mode == 2) {
                    ((float*)outp)[(size_t)i * CC + n] = val;
                } else {
                    const int b = i >> 11;          // /T
                    const int t = i & (TT - 1);
                    const int h = n >> 6;           // /D
                    const int d = n & (DD - 1);
                    size_t idx;
                    if (mode == 0)
                        idx = (((size_t)(b * HH + h)) * TT + t) * DD + d;
                    else
                        idx = (((size_t)(b * HH + h)) * DD + d) * TT + t;
                    ((u16*)outp)[idx] = f2bf(val);
                }
            }
        }
    }
}

// ---------------------------------------------------------------------------
// Kernel 4: flash-style block-causal attention.
//   grid.x = T/16 query blocks, grid.y = B*H, one wave per block.
//   Q,K in [B,H,T,D] bf16; V in [B,H,D,T] bf16; y out [B,T,C] bf16.
// ---------------------------------------------------------------------------
__global__ __launch_bounds__(32) void csa_attn(const u16* __restrict__ Qb,
                                               const u16* __restrict__ Kb,
                                               const u16* __restrict__ Vt,
                                               const int* __restrict__ maskp,
                                               u16* __restrict__ yb) {
    __shared__ alignas(16) u16 ldsP[16 * 32];   // P tile: 16 queries x 32 keys

    const int lane = threadIdx.x;
    const int hf   = lane >> 4;
    const int l15  = lane & 15;
    const int bh   = blockIdx.y;          // b*H + h
    const int b    = bh >> 3;
    const int h    = bh & (HH - 1);
    const int q0   = blockIdx.x * 16;

    const u16* Qp = Qb + (size_t)bh * TT * DD;
    const u16* Kp = Kb + (size_t)bh * TT * DD;
    const u16* Vp = Vt + (size_t)bh * DD * TT;

    // Q fragments: A layout, row t = q0 + l15, two K-chunks of d (0..31, 32..63)
    v16bf qf[2];
#pragma unroll
    for (int kk = 0; kk < 2; ++kk)
        qf[kk] = load_afrag(Qp + (size_t)(q0 + l15) * DD + kk * 32 + hf * 8);

    v8f y[4];
#pragma unroll
    for (int jt = 0; jt < 4; ++jt) y[jt] = zero8();

    float m_r[8], l_r[8];
#pragma unroll
    for (int r = 0; r < 8; ++r) { m_r[r] = -3.0e38f; l_r[r] = 0.0f; }

    const int kend = q0 + 16;   // exclusive bound on keys visible to this block
    const float scale = 0.125f; // 1/sqrt(64)

    for (int k0 = 0; k0 < kend; k0 += 32) {
        // prefetch next key chunk (global_prefetch_b8)
        if (k0 + 32 < kend)
            __builtin_prefetch(Kp + (size_t)(k0 + 32 + l15) * DD, 0, 0);

        // ---- S = Q K^T (two 16-wide key tiles), scale + block-causal mask
        v8f s[2];
#pragma unroll
        for (int j = 0; j < 2; ++j) {
            const int tk = k0 + j * 16 + l15;           // this lane's key column
            v8f sj = zero8();
#pragma unroll
            for (int kk = 0; kk < 2; ++kk) {
                v16bf kf = load_bfrag(Kp + (size_t)tk * DD + kk * 32 + hf * 8);
                sj = wmma_bf16(qf[kk], kf, sj);
            }
            const int mv = maskp[b * TT + tk];
#pragma unroll
            for (int r = 0; r < 8; ++r) {
                const int q = q0 + hf * 8 + r;
                const bool ok = ((tk >> 2) <= (q >> 2)) && (mv != 0);
                sj[r] = ok ? sj[r] * scale : -3.0e38f;
            }
            s[j] = sj;
        }

        // ---- online softmax update (row stats live per-half, 8 rows each)
        float alpha[8];
#pragma unroll
        for (int r = 0; r < 8; ++r) {
            float cm = redmax16(fmaxf(s[0][r], s[1][r]));
            float nm = fmaxf(m_r[r], cm);
            alpha[r] = __expf(m_r[r] - nm);
            m_r[r]   = nm;
        }
        float rs[8];
#pragma unroll
        for (int r = 0; r < 8; ++r) rs[r] = 0.0f;
#pragma unroll
        for (int j = 0; j < 2; ++j) {
#pragma unroll
            for (int r = 0; r < 8; ++r) {
                float p = __expf(s[j][r] - m_r[r]);
                rs[r] += p;
                // store P tile to LDS in row-major (row m = hf*8+r)
                ldsP[(hf * 8 + r) * 32 + j * 16 + l15] = f2bf(p);
            }
        }
#pragma unroll
        for (int r = 0; r < 8; ++r) {
            float cs = redsum16(rs[r]);
            l_r[r] = l_r[r] * alpha[r] + cs;
#pragma unroll
            for (int jt = 0; jt < 4; ++jt) y[jt][r] *= alpha[r];
        }

        __syncthreads();   // LDS visibility across the two half-wave writers

        // ---- re-load P as an A fragment (row m = l15, 32 keys)
        Frag pf;
        {
            const char* base = (const char*)ldsP + l15 * 64 + hf * 16;
            pf.u[0] = *reinterpret_cast<const v4u*>(base);
            pf.u[1] = *reinterpret_cast<const v4u*>(base + 32);
        }

        // ---- Y += P * V  (V stored [d][t]: contiguous over keys)
#pragma unroll
        for (int jt = 0; jt < 4; ++jt) {
            const u16* vp = Vp + (size_t)(jt * 16 + l15) * TT + k0 + hf * 8;
            v16bf vf = load_bfrag(vp);
            y[jt] = wmma_bf16(pf.v, vf, y[jt]);
        }

        __syncthreads();   // protect ldsP before next chunk overwrites it
    }

    // ---- normalize and store y as bf16 [B, T, C] (heads interleaved)
#pragma unroll
    for (int jt = 0; jt < 4; ++jt) {
#pragma unroll
        for (int r = 0; r < 8; ++r) {
            const int t = q0 + hf * 8 + r;
            const int n = h * DD + jt * 16 + l15;
            yb[((size_t)(b * TT + t)) * CC + n] = f2bf(y[jt][r] / l_r[r]);
        }
    }
}

// ---------------------------------------------------------------------------
// Host-side orchestration
// ---------------------------------------------------------------------------
extern "C" void kernel_launch(void* const* d_in, const int* in_sizes, int n_in,
                              void* d_out, int out_size, void* d_ws, size_t ws_size,
                              hipStream_t stream) {
    (void)in_sizes; (void)n_in; (void)out_size; (void)ws_size;

    const float* x    = (const float*)d_in[0];
    const int*   mask = (const int*)  d_in[1];
    const float* Wq   = (const float*)d_in[2];
    const float* bq   = (const float*)d_in[3];
    const float* Wk   = (const float*)d_in[4];
    const float* bk   = (const float*)d_in[5];
    const float* Wv   = (const float*)d_in[6];
    const float* bv   = (const float*)d_in[7];
    const float* Wp   = (const float*)d_in[8];
    const float* bp   = (const float*)d_in[9];

    // workspace layout (bf16 as u16), ~22 MB total
    u16* ws  = (u16*)d_ws;
    u16* xb  = ws;                         // M*C        = 2,097,152
    u16* WqT = xb  + (size_t)MM * CC;      // C*C        =   262,144
    u16* WkT = WqT + (size_t)CC * CC;
    u16* WvT = WkT + (size_t)CC * CC;
    u16* WpT = WvT + (size_t)CC * CC;
    u16* Qb  = WpT + (size_t)CC * CC;      // B*H*T*D    = 2,097,152
    u16* Kb  = Qb  + (size_t)BB * HH * TT * DD;
    u16* Vt  = Kb  + (size_t)BB * HH * TT * DD;
    u16* yb  = Vt  + (size_t)BB * HH * TT * DD;

    // 1) convert x to bf16
    csa_cvt_bf16<<<(MM * CC) / 256, 256, 0, stream>>>(x, xb, MM * CC);

    // 2) convert + transpose weights
    csa_wT_bf16<<<(CC * CC) / 256, 256, 0, stream>>>(Wq, WqT);
    csa_wT_bf16<<<(CC * CC) / 256, 256, 0, stream>>>(Wk, WkT);
    csa_wT_bf16<<<(CC * CC) / 256, 256, 0, stream>>>(Wv, WvT);
    csa_wT_bf16<<<(CC * CC) / 256, 256, 0, stream>>>(Wp, WpT);

    // 3) QKV projections (WMMA GEMM, async-LDS weight staging), V transposed
    dim3 ggrid(MM / 64, CC / 64, 1);
    csa_gemm64<<<ggrid, 32, 0, stream>>>(xb, WqT, bq, Qb, 0);
    csa_gemm64<<<ggrid, 32, 0, stream>>>(xb, WkT, bk, Kb, 0);
    csa_gemm64<<<ggrid, 32, 0, stream>>>(xb, WvT, bv, Vt, 1);

    // 4) block-causal flash attention
    csa_attn<<<dim3(TT / 16, BB * HH, 1), 32, 0, stream>>>(Qb, Kb, Vt, mask, yb);

    // 5) output projection straight into d_out (f32)
    csa_gemm64<<<ggrid, 32, 0, stream>>>(yb, WpT, bp, d_out, 2);
}
